// Social_Aggregator_30039001268869
// MI455X (gfx1250) — matrix operand order repaired and checked by reading, not compiled, not run on hardware
//
#include <hip/hip_runtime.h>
#include <hip/hip_bf16.h>

// CDNA5 (gfx1250) wave32 WMMA implementation of the social-aggregator
// attention layer. Memory-bound by the random e_u gather (~328MB @ 23.3TB/s);
// matmuls run as v_wmma_f32_16x16x32_bf16 with fp32 accumulation.
// Center-embedding half of GEMM1 is folded into an effective bias (halves
// GEMM1 work); b3 is dropped (constant logit shift cancels in softmax).

typedef __attribute__((ext_vector_type(16))) __bf16 v16bf;
typedef __attribute__((ext_vector_type(8)))  float  v8f;

#define DD      128
#define DEG     32
#define PAD_ROW 144   // ushorts per LDS bf16 row: 72 dwords -> conflict-free strided b128 reads

__device__ __forceinline__ unsigned short f2bf(float f) {
  unsigned int u = __float_as_uint(f);
  u += 0x7fffu + ((u >> 16) & 1u);            // round-to-nearest-even
  return (unsigned short)(u >> 16);
}
__device__ __forceinline__ float bf2f(unsigned short h) {
  return __uint_as_float(((unsigned int)h) << 16);
}

union FragBF { uint4 q[2]; v16bf v; };

// ---- weight prep: fp32 row-major -> bf16 transposed (N-major, contiguous K) ----
__global__ void prep_weights_kernel(const float* __restrict__ W1,
                                    const float* __restrict__ W2,
                                    unsigned short* __restrict__ WT1,   // [128][256]
                                    unsigned short* __restrict__ WT2) { // [128][128]
  int idx = blockIdx.x * blockDim.x + threadIdx.x;
  if (idx < 128 * 256) {
    int d = idx >> 8, k = idx & 255;
    WT1[idx] = f2bf(W1[k * 128 + d]);         // WT1[d][k] = W1[k][d]
  } else {
    int j = idx - 128 * 256;
    if (j < 128 * 128) {
      int d = j >> 7, k = j & 127;
      WT2[j] = f2bf(W2[k * 128 + d]);
    }
  }
}

__global__ __launch_bounds__(256)
void social_agg_kernel(const int*   __restrict__ nodes,
                       const int*   __restrict__ neigh_idx,
                       const float* __restrict__ u2e,
                       const float* __restrict__ b1,
                       const float* __restrict__ b2,
                       const float* __restrict__ w3,
                       const unsigned short* __restrict__ WT1,
                       const unsigned short* __restrict__ WT2,
                       float* __restrict__ out) {
  __shared__ unsigned short EUbf[DEG * PAD_ROW];  // A-matrix for GEMM1 (bf16)
  __shared__ unsigned short H1bf[DEG * PAD_ROW];  // A-matrix for GEMM2 (bf16)
  __shared__ float EUf[DEG * DD];                 // fp32 e_u for final weighted sum
  __shared__ float ucf[DD];                       // center embedding
  __shared__ float biasef[DD];                    // b1 + u_c @ W1[128:,:]
  __shared__ float sarr[DEG];                     // attention logits
  __shared__ float attarr[DEG];                   // softmax weights

  const int i    = blockIdx.x;
  const int t    = threadIdx.x;
  const int lane = t & 31;
  const int w    = t >> 5;
  const int lo   = lane & 15;
  const int hi   = lane >> 4;
  const int m    = w & 1;       // M-tile (rows 16m..16m+15)
  const int n0   = w >> 1;      // this wave's N-tiles: n0 and n0+4

  // ---- Phase A: gather neighbor + center embeddings (the HBM-bound part) ----
  {
    int e = t >> 3;             // edge 0..31, 8 threads per edge
    int p = t & 7;              // 16-float chunk within the row
    int ni = neigh_idx[i * DEG + e];
    const float* src = u2e + (size_t)ni * DD + p * 16;
    unsigned int pk[8];
#pragma unroll
    for (int j = 0; j < 4; ++j) {
      float4 v = *(const float4*)(src + 4 * j);
      *(float4*)&EUf[e * DD + p * 16 + 4 * j] = v;              // ds_write_b128
      pk[2 * j + 0] = (unsigned)f2bf(v.x) | ((unsigned)f2bf(v.y) << 16);
      pk[2 * j + 1] = (unsigned)f2bf(v.z) | ((unsigned)f2bf(v.w) << 16);
    }
    uint4* dstq = (uint4*)&EUbf[e * PAD_ROW + p * 16];          // 16B-aligned
    dstq[0] = make_uint4(pk[0], pk[1], pk[2], pk[3]);           // ds_write_b128
    dstq[1] = make_uint4(pk[4], pk[5], pk[6], pk[7]);           // ds_write_b128
    if (t < DEG) {
      float4 v = *(const float4*)(u2e + (size_t)nodes[i] * DD + t * 4);
      *(float4*)&ucf[t * 4] = v;
      sarr[t] = 0.0f;
    }
  }
  __syncthreads();

  // ---- Phase B: effective bias = b1 + u_center @ W1[128:256,:] (once per node) ----
  if (t < DD) {
    const unsigned short* base = WT1 + t * 256 + 128;  // center half of column t
    float sum = b1[t];
#pragma unroll
    for (int k0 = 0; k0 < 128; k0 += 8) {
      uint4 q = *(const uint4*)(base + k0);
      unsigned int qa[4] = {q.x, q.y, q.z, q.w};
#pragma unroll
      for (int j = 0; j < 4; ++j) {
        sum += ucf[k0 + 2 * j + 0] * bf2f((unsigned short)(qa[j] & 0xffffu));
        sum += ucf[k0 + 2 * j + 1] * bf2f((unsigned short)(qa[j] >> 16));
      }
    }
    biasef[t] = sum;
  }
  __syncthreads();

  const v8f vzero = {0, 0, 0, 0, 0, 0, 0, 0};

  // ---- Phase C: GEMM1  H1 = relu(EU @ W1[:128,:] + bias_eff)  (K = 128) ----
  {
    v8f acc0 = vzero, acc1 = vzero;
#pragma unroll
    for (int kk = 0; kk < 4; ++kk) {
      // A frag (16x32 bf16): lane holds row lo, K = kk*32 + 8*hi + {0..7, 16..23}
      const unsigned short* ap = &EUbf[(m * 16 + lo) * PAD_ROW + kk * 32 + hi * 8];
      FragBF A; A.q[0] = *(const uint4*)ap; A.q[1] = *(const uint4*)(ap + 16);
      // B frags (32x16 bf16): lane holds col lo, K = kk*32 + 8*hi + {0..7, 16..23}
      const unsigned short* bp0 = WT1 + (n0 * 16 + lo) * 256 + kk * 32 + hi * 8;
      FragBF B0; B0.q[0] = *(const uint4*)bp0; B0.q[1] = *(const uint4*)(bp0 + 16);
      const unsigned short* bp1 = WT1 + ((n0 + 4) * 16 + lo) * 256 + kk * 32 + hi * 8;
      FragBF B1; B1.q[0] = *(const uint4*)bp1; B1.q[1] = *(const uint4*)(bp1 + 16);
      acc0 = __builtin_amdgcn_wmma_f32_16x16x32_bf16(false, A.v, false, B0.v,
                                                     (short)0, acc0, false, false);
      acc1 = __builtin_amdgcn_wmma_f32_16x16x32_bf16(false, A.v, false, B1.v,
                                                     (short)0, acc1, false, false);
    }
    int d0 = n0 * 16 + lo, d1 = (n0 + 4) * 16 + lo;
    float be0 = biasef[d0], be1 = biasef[d1];
#pragma unroll
    for (int v = 0; v < 8; ++v) {            // C layout: M = 16m + 8*hi + v, N = lo
      int r = m * 16 + hi * 8 + v;
      H1bf[r * PAD_ROW + d0] = f2bf(fmaxf(acc0[v] + be0, 0.0f));
      H1bf[r * PAD_ROW + d1] = f2bf(fmaxf(acc1[v] + be1, 0.0f));
    }
  }
  __syncthreads();

  // ---- Phase D: GEMM2 + fused logit  s = relu(H1 @ W2 + b2) @ w3  (K = 128) ----
  {
    v8f acc0 = vzero, acc1 = vzero;
#pragma unroll
    for (int kk = 0; kk < 4; ++kk) {
      const unsigned short* ap = &H1bf[(m * 16 + lo) * PAD_ROW + kk * 32 + hi * 8];
      FragBF A; A.q[0] = *(const uint4*)ap; A.q[1] = *(const uint4*)(ap + 16);
      const unsigned short* bp0 = WT2 + (n0 * 16 + lo) * 128 + kk * 32 + hi * 8;
      FragBF B0; B0.q[0] = *(const uint4*)bp0; B0.q[1] = *(const uint4*)(bp0 + 16);
      const unsigned short* bp1 = WT2 + ((n0 + 4) * 16 + lo) * 128 + kk * 32 + hi * 8;
      FragBF B1; B1.q[0] = *(const uint4*)bp1; B1.q[1] = *(const uint4*)(bp1 + 16);
      acc0 = __builtin_amdgcn_wmma_f32_16x16x32_bf16(false, A.v, false, B0.v,
                                                     (short)0, acc0, false, false);
      acc1 = __builtin_amdgcn_wmma_f32_16x16x32_bf16(false, A.v, false, B1.v,
                                                     (short)0, acc1, false, false);
    }
    int d0 = n0 * 16 + lo, d1 = (n0 + 4) * 16 + lo;
    float b20 = b2[d0], w30 = w3[d0];
    float b21 = b2[d1], w31 = w3[d1];
    float part[8];
#pragma unroll
    for (int v = 0; v < 8; ++v)
      part[v] = fmaxf(acc0[v] + b20, 0.0f) * w30 +
                fmaxf(acc1[v] + b21, 0.0f) * w31;
    // reduce over N within each 16-lane half (rows are constant per half-wave)
#pragma unroll
    for (int off = 1; off < 16; off <<= 1)
#pragma unroll
      for (int v = 0; v < 8; ++v)
        part[v] += __shfl_xor(part[v], off, 16);
    if (lo == 0) {
#pragma unroll
      for (int v = 0; v < 8; ++v)
        atomicAdd(&sarr[m * 16 + hi * 8 + v], part[v]);   // ds_add_f32
    }
  }
  __syncthreads();

  // ---- Phase E: softmax over the 32 neighbors (wave 0; b3 cancels) ----
  if (w == 0) {
    float s  = sarr[lane];
    float mx = s;
#pragma unroll
    for (int off = 16; off; off >>= 1) mx = fmaxf(mx, __shfl_xor(mx, off, 32));
    float ex = __expf(s - mx);
    float tot = ex;
#pragma unroll
    for (int off = 16; off; off >>= 1) tot += __shfl_xor(tot, off, 32);
    attarr[lane] = ex / tot;
  }
  __syncthreads();

  // ---- Phase F: out[i] = sum_e att[e] * e_u[e]  (fp32) ----
  if (t < DD) {
    float acc = 0.0f;
#pragma unroll 8
    for (int e = 0; e < DEG; ++e) acc += attarr[e] * EUf[e * DD + t];
    out[(size_t)i * DD + t] = acc;
  }
}

extern "C" void kernel_launch(void* const* d_in, const int* in_sizes, int n_in,
                              void* d_out, int out_size, void* d_ws, size_t ws_size,
                              hipStream_t stream) {
  const int*   nodes = (const int*)d_in[0];
  const int*   neigh = (const int*)d_in[1];
  // d_in[2] seg_ids unused: fixed-degree layout => edges of node i are [32i, 32i+32)
  const float* u2e   = (const float*)d_in[3];
  const float* W1    = (const float*)d_in[4];
  const float* b1    = (const float*)d_in[5];
  const float* W2    = (const float*)d_in[6];
  const float* b2    = (const float*)d_in[7];
  const float* w3    = (const float*)d_in[8];
  // d_in[9] b3 unused: uniform logit shift cancels in segment softmax

  unsigned short* WT1 = (unsigned short*)d_ws;      // 128*256 bf16 = 64KB
  unsigned short* WT2 = WT1 + 128 * 256;            // 128*128 bf16 = 32KB

  const int n_nodes = in_sizes[0];

  prep_weights_kernel<<<192, 256, 0, stream>>>(W1, W2, WT1, WT2);
  social_agg_kernel<<<n_nodes, 256, 0, stream>>>(nodes, neigh, u2e, b1, b2, w3,
                                                 WT1, WT2, (float*)d_out);
}
